// triton_ag_gemm_intra_node_54563264529072
// MI455X (gfx1250) — matrix operand order repaired and checked by reading, not compiled, not run on hardware
//
#include <hip/hip_runtime.h>

// C[M,N] = A[M,K] * W[N,K]^T in fp32-equivalent precision via bf16x3 split
// (A_hi*B_hi + A_hi*B_lo + A_lo*B_hi) on v_wmma_f32_16x16x32_bf16.
//
// Two-pass: pre-split A/W into bf16 hi/lo planes in d_ws (memory-bound),
// then a conversion-free GEMM that stages planes into double-buffered LDS with
// gfx1250 async global->LDS copies (ASYNCcnt) when the builtin is available.
// Falls back to a fused split kernel if d_ws is too small.

namespace {

constexpr int Mdim = 4096;
constexpr int Kdim = 8192;
constexpr int Ndim = 8192;

constexpr int BM = 128;
constexpr int BN = 128;
constexpr int BK = 32;
constexpr int LSTR = BK + 8;            // LDS row stride in bf16 elems (80 B rows)
constexpr int TILE_ELEMS = BM * LSTR;

typedef __attribute__((ext_vector_type(16))) __bf16          v16bf;
typedef __attribute__((ext_vector_type(8)))  float           v8f;
typedef __attribute__((ext_vector_type(4)))  float           v4f;
typedef __attribute__((ext_vector_type(4)))  unsigned int    v4u;
typedef __attribute__((ext_vector_type(8)))  unsigned short  v8us;
typedef int v4i_vs __attribute__((vector_size(16)));         // matches builtin param

union Frag {
  v16bf bf;
  v4u   u[2];
};

#if __has_builtin(__builtin_amdgcn_global_load_async_to_lds_b128)
#define USE_ASYNC_LDS 1
#else
#define USE_ASYNC_LDS 0
#endif

#if USE_ASYNC_LDS
#define ASYNC_B128(gsrc, ldst)                                              \
  __builtin_amdgcn_global_load_async_to_lds_b128(                           \
      (__attribute__((address_space(1))) v4i_vs*)(gsrc),                    \
      (__attribute__((address_space(3))) v4i_vs*)(ldst), 0, 0)
#if __has_builtin(__builtin_amdgcn_s_wait_asynccnt)
#define WAIT_ASYNC() __builtin_amdgcn_s_wait_asynccnt(0)
#else
#define WAIT_ASYNC() asm volatile("s_wait_asynccnt 0" ::: "memory")
#endif
#endif

// A-operand fragment (16-bit A 16x32 layout):
// lanes 0-15 : K 0-7 at +0, K16-23 at +16 elems ; lanes 16-31 : K 8-15, K 24-31.
__device__ __forceinline__ v16bf ld_fragA(const unsigned short* t, int row, int sel) {
  const unsigned short* p = t + row * LSTR + sel * 8;
  Frag f;
  f.u[0] = *(const v4u*)(p);
  f.u[1] = *(const v4u*)(p + 16);
  return f.bf;
}

// B-operand fragment (16-bit B 32x16 layout, N striped across lanes):
// lanes 0-15 : K 0-7 at +0, K16-23 at +16 elems ; lanes 16-31 : K 8-15, K 24-31.
__device__ __forceinline__ v16bf ld_fragB(const unsigned short* t, int row, int sel) {
  const unsigned short* p = t + row * LSTR + sel * 8;
  Frag f;
  f.u[0] = *(const v4u*)(p);
  f.u[1] = *(const v4u*)(p + 16);
  return f.bf;
}

// Split 8 fp32 into bf16 hi (truncation) + bf16 lo (residual).
__device__ __forceinline__ void split8(const v4f f0, const v4f f1, v8us& h, v8us& l) {
#pragma unroll
  for (int i = 0; i < 8; ++i) {
    float x = (i < 4) ? f0[i] : f1[i - 4];
    unsigned int b = __float_as_uint(x);
    h[i] = (unsigned short)(b >> 16);
    float hf = __uint_as_float(b & 0xFFFF0000u);
    float lo = x - hf;
    l[i] = (unsigned short)(__float_as_uint(lo) >> 16);
  }
}

// Supertile swizzle: 16x16 blocks per supertile -> 64MB(A)+64MB(W)=128MB < 192MB L2.
__device__ __forceinline__ void tile_coords(int bid, int& mBase, int& nBase) {
  constexpr int MB  = Mdim / BM;   // 32
  constexpr int GM  = 16, GN = 16;
  constexpr int SM  = MB / GM;     // 2 supertile rows
  constexpr int PER = GM * GN;     // 256 blocks per supertile
  const int st  = bid / PER;
  const int r   = bid % PER;
  const int stm = st % SM;
  const int stn = st / SM;
  mBase = (stm * GM + (r % GM)) * BM;
  nBase = (stn * GN + (r / GM)) * BN;
}

} // namespace

// ---------------------------------------------------------------------------
// Pass 1: elementwise fp32 -> (bf16 hi, bf16 lo) planes. 8 floats per thread.
// ---------------------------------------------------------------------------
__global__ __launch_bounds__(256) void split_planes(
    const float* __restrict__ src,
    unsigned short* __restrict__ hi,
    unsigned short* __restrict__ lo)
{
  const size_t i = ((size_t)blockIdx.x * 256 + threadIdx.x) * 8;
  const v4f f0 = *(const v4f*)(src + i);
  const v4f f1 = *(const v4f*)(src + i + 4);
  v8us h, l;
  split8(f0, f1, h, l);
  *(v8us*)(hi + i) = h;
  *(v8us*)(lo + i) = l;
}

// ---------------------------------------------------------------------------
// Pass 2: conversion-free bf16x3 WMMA GEMM from pre-split planes.
// ---------------------------------------------------------------------------
__global__ __launch_bounds__(256) void ag_gemm_bf16x3_pre(
    const unsigned short* __restrict__ AHg,  // [M, K] bf16 hi
    const unsigned short* __restrict__ ALg,  // [M, K] bf16 lo
    const unsigned short* __restrict__ WHg,  // [N, K] bf16 hi
    const unsigned short* __restrict__ WLg,  // [N, K] bf16 lo
    float* __restrict__ C)                   // [M, N]
{
  __shared__ unsigned short AH[2][TILE_ELEMS];
  __shared__ unsigned short AL[2][TILE_ELEMS];
  __shared__ unsigned short BH[2][TILE_ELEMS];
  __shared__ unsigned short BL[2][TILE_ELEMS];

  const int tid  = threadIdx.x;
  const int lane = tid & 31;
  const int wid  = tid >> 5;
  const int wm   = wid & 1;    // wave M index (0..1), 64 rows each
  const int wn   = wid >> 1;   // wave N index (0..3), 32 cols each
  const int lrow = lane & 15;
  const int sel  = lane >> 4;

  int mBase, nBase;
  tile_coords(blockIdx.x, mBase, nBase);

  v8f acc[8];
#pragma unroll
  for (int t = 0; t < 8; ++t)
#pragma unroll
    for (int j = 0; j < 8; ++j) acc[t][j] = 0.0f;

#if !USE_ASYNC_LDS
  v8us sa[2][2], sb[2][2];   // [plane hi/lo][pass] staging regs
#endif

  // Stage one 128x32 hi+lo tile pair of A and W for K offset k0 into buffer.
  auto stage_issue = [&](int buf, int k0) {
#pragma unroll
    for (int p = 0; p < 2; ++p) {
      const int c   = p * 256 + tid;       // 512 chunks of 8 bf16 per plane-tile
      const int r   = c >> 2;
      const int col = (c & 3) << 3;
      const size_t ga = (size_t)(mBase + r) * Kdim + k0 + col;
      const size_t gb = (size_t)(nBase + r) * Kdim + k0 + col;
      const int    la = r * LSTR + col;
#if USE_ASYNC_LDS
      ASYNC_B128(AHg + ga, &AH[buf][la]);
      ASYNC_B128(ALg + ga, &AL[buf][la]);
      ASYNC_B128(WHg + gb, &BH[buf][la]);
      ASYNC_B128(WLg + gb, &BL[buf][la]);
#else
      sa[0][p] = *(const v8us*)(AHg + ga);
      sa[1][p] = *(const v8us*)(ALg + ga);
      sb[0][p] = *(const v8us*)(WHg + gb);
      sb[1][p] = *(const v8us*)(WLg + gb);
      (void)buf; (void)la;
#endif
    }
  };

#if !USE_ASYNC_LDS
  auto stage_commit = [&](int buf) {
#pragma unroll
    for (int p = 0; p < 2; ++p) {
      const int c   = p * 256 + tid;
      const int r   = c >> 2;
      const int col = (c & 3) << 3;
      const int la  = r * LSTR + col;
      *(v8us*)&AH[buf][la] = sa[0][p];
      *(v8us*)&AL[buf][la] = sa[1][p];
      *(v8us*)&BH[buf][la] = sb[0][p];
      *(v8us*)&BL[buf][la] = sb[1][p];
    }
  };
#endif

  auto compute = [&](int buf) {
    v16bf aH[4], aL[4], bH[2], bL[2];
#pragma unroll
    for (int mt = 0; mt < 4; ++mt) {
      const int r = wm * 64 + mt * 16 + lrow;
      aH[mt] = ld_fragA(AH[buf], r, sel);
      aL[mt] = ld_fragA(AL[buf], r, sel);
    }
#pragma unroll
    for (int nt = 0; nt < 2; ++nt) {
      const int r = wn * 32 + nt * 16 + lrow;
      bH[nt] = ld_fragB(BH[buf], r, sel);
      bL[nt] = ld_fragB(BL[buf], r, sel);
    }
    // Three product passes; each accumulator is revisited only after 7 other
    // WMMAs, avoiding back-to-back WMMA->WMMA RAW hazard stalls.
#pragma unroll
    for (int mt = 0; mt < 4; ++mt)
#pragma unroll
      for (int nt = 0; nt < 2; ++nt) {
        const int t = mt * 2 + nt;
        acc[t] = __builtin_amdgcn_wmma_f32_16x16x32_bf16(
            false, aH[mt], false, bH[nt], (short)0, acc[t], false, false);
      }
#pragma unroll
    for (int mt = 0; mt < 4; ++mt)
#pragma unroll
      for (int nt = 0; nt < 2; ++nt) {
        const int t = mt * 2 + nt;
        acc[t] = __builtin_amdgcn_wmma_f32_16x16x32_bf16(
            false, aH[mt], false, bL[nt], (short)0, acc[t], false, false);
      }
#pragma unroll
    for (int mt = 0; mt < 4; ++mt)
#pragma unroll
      for (int nt = 0; nt < 2; ++nt) {
        const int t = mt * 2 + nt;
        acc[t] = __builtin_amdgcn_wmma_f32_16x16x32_bf16(
            false, aL[mt], false, bH[nt], (short)0, acc[t], false, false);
      }
  };

  // Prologue: stage K-tile 0 into buffer 0.
#if USE_ASYNC_LDS
  stage_issue(0, 0);
  WAIT_ASYNC();
#else
  stage_issue(0, 0);
  stage_commit(0);
#endif
  __syncthreads();

  int buf = 0;
  for (int kb = 1; kb < Kdim / BK; ++kb) {
    const int k0 = kb * BK;
#if USE_ASYNC_LDS
    stage_issue(buf ^ 1, k0);   // async copies overlap the WMMAs below
    compute(buf);
    WAIT_ASYNC();
#else
    stage_issue(buf ^ 1, k0);   // global loads overlap the WMMAs below
    compute(buf);
    stage_commit(buf ^ 1);
#endif
    __syncthreads();
    buf ^= 1;
  }
  compute(buf);

  // Epilogue: C/D 16x16 f32 layout — VGPR r: lanes 0-15 M=r, lanes 16-31 M=8+r.
#pragma unroll
  for (int mt = 0; mt < 4; ++mt)
#pragma unroll
    for (int nt = 0; nt < 2; ++nt) {
      const v8f c = acc[mt * 2 + nt];
      const int gm = mBase + wm * 64 + mt * 16 + sel * 8;
      const int gn = nBase + wn * 32 + nt * 16 + lrow;
#pragma unroll
      for (int r = 0; r < 8; ++r)
        C[(size_t)(gm + r) * Ndim + gn] = c[r];
    }
}

// ---------------------------------------------------------------------------
// Fallback: fused split + GEMM (used only if d_ws cannot hold the planes).
// ---------------------------------------------------------------------------
__global__ __launch_bounds__(256) void ag_gemm_bf16x3_fused(
    const float* __restrict__ A,   // [M, K]
    const float* __restrict__ W,   // [N, K]
    float* __restrict__ C)         // [M, N]
{
  __shared__ unsigned short AH[2][TILE_ELEMS];
  __shared__ unsigned short AL[2][TILE_ELEMS];
  __shared__ unsigned short BH[2][TILE_ELEMS];
  __shared__ unsigned short BL[2][TILE_ELEMS];

  const int tid  = threadIdx.x;
  const int lane = tid & 31;
  const int wid  = tid >> 5;
  const int wm   = wid & 1;
  const int wn   = wid >> 1;
  const int lrow = lane & 15;
  const int sel  = lane >> 4;

  int mBase, nBase;
  tile_coords(blockIdx.x, mBase, nBase);

  v8f acc[8];
#pragma unroll
  for (int t = 0; t < 8; ++t)
#pragma unroll
    for (int j = 0; j < 8; ++j) acc[t][j] = 0.0f;

  v4f ra0[2], ra1[2], rb0[2], rb1[2];

  auto load_regs = [&](int k0) {
#pragma unroll
    for (int p = 0; p < 2; ++p) {
      const int c   = p * 256 + tid;
      const int r   = c >> 2;
      const int col = (c & 3) << 3;
      const float* ga = A + (size_t)(mBase + r) * Kdim + k0 + col;
      const float* gb = W + (size_t)(nBase + r) * Kdim + k0 + col;
      ra0[p] = *(const v4f*)(ga);
      ra1[p] = *(const v4f*)(ga + 4);
      rb0[p] = *(const v4f*)(gb);
      rb1[p] = *(const v4f*)(gb + 4);
    }
  };

  auto store_stage = [&](int buf) {
#pragma unroll
    for (int p = 0; p < 2; ++p) {
      const int c   = p * 256 + tid;
      const int r   = c >> 2;
      const int col = (c & 3) << 3;
      const int la  = r * LSTR + col;
      v8us h, l;
      split8(ra0[p], ra1[p], h, l);
      *(v8us*)&AH[buf][la] = h;
      *(v8us*)&AL[buf][la] = l;
      split8(rb0[p], rb1[p], h, l);
      *(v8us*)&BH[buf][la] = h;
      *(v8us*)&BL[buf][la] = l;
    }
  };

  auto compute = [&](int buf) {
    v16bf aH[4], aL[4], bH[2], bL[2];
#pragma unroll
    for (int mt = 0; mt < 4; ++mt) {
      const int r = wm * 64 + mt * 16 + lrow;
      aH[mt] = ld_fragA(AH[buf], r, sel);
      aL[mt] = ld_fragA(AL[buf], r, sel);
    }
#pragma unroll
    for (int nt = 0; nt < 2; ++nt) {
      const int r = wn * 32 + nt * 16 + lrow;
      bH[nt] = ld_fragB(BH[buf], r, sel);
      bL[nt] = ld_fragB(BL[buf], r, sel);
    }
#pragma unroll
    for (int mt = 0; mt < 4; ++mt)
#pragma unroll
      for (int nt = 0; nt < 2; ++nt) {
        const int t = mt * 2 + nt;
        acc[t] = __builtin_amdgcn_wmma_f32_16x16x32_bf16(
            false, aH[mt], false, bH[nt], (short)0, acc[t], false, false);
      }
#pragma unroll
    for (int mt = 0; mt < 4; ++mt)
#pragma unroll
      for (int nt = 0; nt < 2; ++nt) {
        const int t = mt * 2 + nt;
        acc[t] = __builtin_amdgcn_wmma_f32_16x16x32_bf16(
            false, aH[mt], false, bL[nt], (short)0, acc[t], false, false);
      }
#pragma unroll
    for (int mt = 0; mt < 4; ++mt)
#pragma unroll
      for (int nt = 0; nt < 2; ++nt) {
        const int t = mt * 2 + nt;
        acc[t] = __builtin_amdgcn_wmma_f32_16x16x32_bf16(
            false, aL[mt], false, bH[nt], (short)0, acc[t], false, false);
      }
  };

  load_regs(0);
  store_stage(0);
  __syncthreads();

  int buf = 0;
  for (int kb = 1; kb < Kdim / BK; ++kb) {
    const int k0 = kb * BK;
    load_regs(k0);
    compute(buf);
    store_stage(buf ^ 1);
    __syncthreads();
    buf ^= 1;
  }
  compute(buf);

#pragma unroll
  for (int mt = 0; mt < 4; ++mt)
#pragma unroll
    for (int nt = 0; nt < 2; ++nt) {
      const v8f c = acc[mt * 2 + nt];
      const int gm = mBase + wm * 64 + mt * 16 + sel * 8;
      const int gn = nBase + wn * 32 + nt * 16 + lrow;
#pragma unroll
      for (int r = 0; r < 8; ++r)
        C[(size_t)(gm + r) * Ndim + gn] = c[r];
    }
}

extern "C" void kernel_launch(void* const* d_in, const int* in_sizes, int n_in,
                              void* d_out, int out_size, void* d_ws, size_t ws_size,
                              hipStream_t stream) {
  (void)in_sizes; (void)n_in; (void)out_size;
  const float* A = (const float*)d_in[0];   // [M, K]
  const float* W = (const float*)d_in[1];   // [N, K]
  float* C = (float*)d_out;                 // [M, N]

  constexpr size_t A_ELEMS = (size_t)Mdim * Kdim;          // 32M
  constexpr size_t W_ELEMS = (size_t)Ndim * Kdim;          // 64M
  constexpr size_t WS_NEED = (A_ELEMS + W_ELEMS) * 2 * sizeof(unsigned short);
  const int nBlocks = (Mdim / BM) * (Ndim / BN);           // 2048

  if (ws_size >= WS_NEED) {
    unsigned short* AHg = (unsigned short*)d_ws;
    unsigned short* ALg = AHg + A_ELEMS;
    unsigned short* WHg = ALg + A_ELEMS;
    unsigned short* WLg = WHg + W_ELEMS;
    split_planes<<<dim3((unsigned)(A_ELEMS / 2048)), dim3(256), 0, stream>>>(A, AHg, ALg);
    split_planes<<<dim3((unsigned)(W_ELEMS / 2048)), dim3(256), 0, stream>>>(W, WHg, WLg);
    ag_gemm_bf16x3_pre<<<dim3(nBlocks), dim3(256), 0, stream>>>(AHg, ALg, WHg, WLg, C);
  } else {
    ag_gemm_bf16x3_fused<<<dim3(nBlocks), dim3(256), 0, stream>>>(A, W, C);
  }
}
